// Decoder_36584531427590
// MI455X (gfx1250) — matrix operand order repaired and checked by reading, not compiled, not run on hardware
//
#include <hip/hip_runtime.h>
#include <math.h>

// ---------------------------------------------------------------------------
// GRU rollout for MI455X (gfx1250, wave32, WMMA).
//   N=16384 rows, HID=256, 3*HID=768 gates, T=96 steps.
//   Block = 256 threads (8 waves), owns 32 batch rows, loops all T steps.
//   Wave w owns hidden-column groups {2w, 2w+1} (16 cols each) and for each
//   the r/z/n gate tiles -> gating is per-lane VALU, no LDS roundtrip.
//   W_hh pre-converted to bf16 in d_ws, streamed from L2 each step.
//   An integer offset (not the pointer) is laundered through asm each step:
//   keeps the loads un-hoistable by LICM *and* keeps addrspace(1) inference
//   so they lower to global_load_b128 (saddr form), not flat_load.
//   A- and B-fragments are double-buffered one K-step ahead for latency hiding.
// ---------------------------------------------------------------------------

typedef __bf16 v16bf __attribute__((ext_vector_type(16)));
typedef float  v8f   __attribute__((ext_vector_type(8)));

#define HID 256
#define CTXD 5
#define HSTRIDE 264   // bf16 elements per LDS row (256 + 8 pad -> conflict-free b128)

__device__ __forceinline__ unsigned short f32_bf16(float f) {
    unsigned u = __builtin_bit_cast(unsigned, f);
    unsigned r = u + 0x7FFFu + ((u >> 16) & 1u);   // round-to-nearest-even
    return (unsigned short)(r >> 16);
}
__device__ __forceinline__ float sigm(float x) { return 1.0f / (1.0f + __expf(-x)); }

__global__ void whh_to_bf16(const float* __restrict__ w, unsigned short* __restrict__ o, int n) {
    int i = blockIdx.x * blockDim.x + threadIdx.x;
    if (i < n) o[i] = f32_bf16(w[i]);
}

__global__ __launch_bounds__(256, 2)
void gru_rollout(const float* __restrict__ init_h, const float* __restrict__ ctx,
                 const float* __restrict__ x0, const float* __restrict__ y0,
                 const float* __restrict__ W_ih, const unsigned short* __restrict__ W_hh_bf,
                 const float* __restrict__ b_ih, const float* __restrict__ b_hh,
                 const float* __restrict__ W_out, const float* __restrict__ b_out,
                 const int* __restrict__ Tp, float* __restrict__ out) {
    __shared__ unsigned short hbuf[32 * HSTRIDE];   // h (bf16), A-matrix staging
    __shared__ float pd[32 * 2];                    // prev_delta per row
    __shared__ float pdsum[32 * 2];                 // W_out partial sums
    __shared__ float ctxs[32 * CTXD];

    const int tid  = threadIdx.x;
    const int lane = tid & 31;
    const int wid  = tid >> 5;
    const int lrow = lane & 15;     // A row / B,C column within tile
    const int lhi  = lane >> 4;     // which half of the wave
    const int rowBase = blockIdx.x * 32;
    const int T = *Tp;
    const float max_step = 10.1415f * 0.093f;

    // column-group cg = gate*2 + t ; gate in {r,z,n}, t picks gtile 2*wid+t
    int colbase[6];
#pragma unroll
    for (int g = 0; g < 3; ++g)
#pragma unroll
        for (int t = 0; t < 2; ++t)
            colbase[g * 2 + t] = g * HID + (2 * wid + t) * 16;

    // per-lane element offsets of the B fragments (K chunk kk added per step)
    int boff[6];
#pragma unroll
    for (int cg = 0; cg < 6; ++cg)
        boff[cg] = (colbase[cg] + lrow) * HID + 16 * lhi;

    if (tid < 32 * CTXD) ctxs[tid] = ctx[(size_t)rowBase * CTXD + tid];
    if (tid < 64) { pd[tid] = 0.0f; pdsum[tid] = 0.0f; }
    __syncthreads();

    // Static input contribution: cc = b_ih (+b_hh for r,z) + ctx @ W_ih[:,2:7].T
    float w0[6], w1[6];
    v8f cc[6][2];
#pragma unroll
    for (int cg = 0; cg < 6; ++cg) {
        int j = colbase[cg] + lrow;
        const float* wr = &W_ih[(size_t)j * 7];
        w0[cg] = wr[0]; w1[cg] = wr[1];
        float c2 = wr[2], c3 = wr[3], c4 = wr[4], c5 = wr[5], c6 = wr[6];
        float base = b_ih[j] + (cg < 4 ? b_hh[j] : 0.0f);
#pragma unroll
        for (int m = 0; m < 2; ++m)
#pragma unroll
            for (int v = 0; v < 8; ++v) {
                const float* cp = &ctxs[(m * 16 + 8 * lhi + v) * CTXD];
                cc[cg][m][v] = base + cp[0]*c2 + cp[1]*c3 + cp[2]*c4 + cp[3]*c5 + cp[4]*c6;
            }
    }
    float bhhn[2] = { b_hh[colbase[4] + lrow], b_hh[colbase[5] + lrow] };
    float wo0[2], wo1[2];
#pragma unroll
    for (int gt = 0; gt < 2; ++gt) {
        int col = (2 * wid + gt) * 16 + lrow;
        wo0[gt] = W_out[col]; wo1[gt] = W_out[HID + col];
    }

    // h: fp32 in registers (elementwise path), bf16 in LDS (matmul operand)
    float hreg[2][2][8];
#pragma unroll
    for (int gt = 0; gt < 2; ++gt) {
        int col = (2 * wid + gt) * 16 + lrow;
#pragma unroll
        for (int m = 0; m < 2; ++m)
#pragma unroll
            for (int v = 0; v < 8; ++v) {
                int row = m * 16 + 8 * lhi + v;
                float h = init_h[(size_t)(rowBase + row) * HID + col];
                hreg[gt][m][v] = h;
                hbuf[row * HSTRIDE + col] = f32_bf16(h);
            }
    }

    float posx = 0.0f, posy = 0.0f, bo0 = 0.0f, bo1 = 0.0f;
    if (wid == 0) {
        posx = x0[rowBase + lane]; posy = y0[rowBase + lane];
        bo0 = b_out[0]; bo1 = b_out[1];
    }
    __syncthreads();

    for (int t = 0; t < T; ++t) {
        // Opaque zero offset, re-made every iteration: loads through
        // (W_hh_bf + ofs) keep global provenance but cannot be hoisted.
        unsigned long long ofs = 0;
        asm volatile("" : "+s"(ofs));
        const unsigned short* whh = W_hh_bf + ofs;

        // prev_delta for this wave's rows
        float dxr[2][8], dyr[2][8];
#pragma unroll
        for (int m = 0; m < 2; ++m)
#pragma unroll
            for (int v = 0; v < 8; ++v) {
                int row = m * 16 + 8 * lhi + v;
                float2 d = *(const float2*)&pd[row * 2];
                dxr[m][v] = d.x; dyr[m][v] = d.y;
            }

        // Init accumulators: r,z tiles get full gi (incl. biases); n tiles get b_hh.
        v8f acc[6][2];
#pragma unroll
        for (int cg = 0; cg < 4; ++cg)
#pragma unroll
            for (int m = 0; m < 2; ++m)
#pragma unroll
                for (int v = 0; v < 8; ++v)
                    acc[cg][m][v] = cc[cg][m][v] + dxr[m][v] * w0[cg] + dyr[m][v] * w1[cg];
#pragma unroll
        for (int gt = 0; gt < 2; ++gt)
#pragma unroll
            for (int m = 0; m < 2; ++m)
#pragma unroll
                for (int v = 0; v < 8; ++v)
                    acc[4 + gt][m][v] = bhhn[gt];

        // --- gh += h @ W_hh.T ---------------------------------------------
        // 8 K-steps, 12 WMMA each; A (LDS) and B (global/L2) fragments are
        // double-buffered one K-step ahead.
        v16bf bcur[6], acur[2];
#pragma unroll
        for (int cg = 0; cg < 6; ++cg)
            bcur[cg] = *(const v16bf*)(whh + boff[cg]);
#pragma unroll
        for (int m = 0; m < 2; ++m) {
            const char* ap = (const char*)hbuf + ((size_t)(m * 16 + lrow) * HSTRIDE + 16 * lhi) * 2;
            ((uint4*)&acur[m])[0] = *(const uint4*)ap;
            ((uint4*)&acur[m])[1] = *(const uint4*)(ap + 16);
        }

#pragma unroll
        for (int kb = 0; kb < 8; ++kb) {
            v16bf bnxt[6], anxt[2];
            if (kb < 7) {
                int kk = (kb + 1) * 32;
#pragma unroll
                for (int cg = 0; cg < 6; ++cg)
                    bnxt[cg] = *(const v16bf*)(whh + boff[cg] + kk);
#pragma unroll
                for (int m = 0; m < 2; ++m) {
                    const char* ap = (const char*)hbuf +
                        ((size_t)(m * 16 + lrow) * HSTRIDE + kk + 16 * lhi) * 2;
                    ((uint4*)&anxt[m])[0] = *(const uint4*)ap;
                    ((uint4*)&anxt[m])[1] = *(const uint4*)(ap + 16);
                }
            }
#pragma unroll
            for (int cg = 0; cg < 6; ++cg) {
                acc[cg][0] = __builtin_amdgcn_wmma_f32_16x16x32_bf16(
                    false, acur[0], false, bcur[cg], (short)0, acc[cg][0], false, false);
                acc[cg][1] = __builtin_amdgcn_wmma_f32_16x16x32_bf16(
                    false, acur[1], false, bcur[cg], (short)0, acc[cg][1], false, false);
            }
            if (kb < 7) {
#pragma unroll
                for (int cg = 0; cg < 6; ++cg) bcur[cg] = bnxt[cg];
                acur[0] = anxt[0]; acur[1] = anxt[1];
            }
        }

        // GRU gating (per-lane: r,z,n tiles share the C layout) + W_out partials
        float px[2][8], py[2][8];
#pragma unroll
        for (int m = 0; m < 2; ++m)
#pragma unroll
            for (int v = 0; v < 8; ++v) { px[m][v] = 0.0f; py[m][v] = 0.0f; }
#pragma unroll
        for (int gt = 0; gt < 2; ++gt)
#pragma unroll
            for (int m = 0; m < 2; ++m)
#pragma unroll
                for (int v = 0; v < 8; ++v) {
                    float r  = sigm(acc[gt][m][v]);
                    float z  = sigm(acc[2 + gt][m][v]);
                    float in = cc[4 + gt][m][v] + dxr[m][v] * w0[4 + gt] + dyr[m][v] * w1[4 + gt];
                    float e  = __expf(2.0f * (in + r * acc[4 + gt][m][v]));
                    float nv = (e - 1.0f) / (e + 1.0f);           // tanh
                    float hn = (1.0f - z) * nv + z * hreg[gt][m][v];
                    hreg[gt][m][v] = hn;
                    px[m][v] += hn * wo0[gt];
                    py[m][v] += hn * wo1[gt];
                }

        __syncthreads();   // all waves done reading hbuf / pd

        // publish h_new (bf16) for next step's A fragments
#pragma unroll
        for (int gt = 0; gt < 2; ++gt) {
            int col = (2 * wid + gt) * 16 + lrow;
#pragma unroll
            for (int m = 0; m < 2; ++m)
#pragma unroll
                for (int v = 0; v < 8; ++v)
                    hbuf[(m * 16 + 8 * lhi + v) * HSTRIDE + col] = f32_bf16(hreg[gt][m][v]);
        }

        // reduce W_out partials across the 16 column-lanes, then LDS atomics
#pragma unroll
        for (int m = 0; m < 2; ++m)
#pragma unroll
            for (int v = 0; v < 8; ++v) {
                float sx = px[m][v], sy = py[m][v];
#pragma unroll
                for (int off = 8; off >= 1; off >>= 1) {
                    sx += __shfl_xor(sx, off, 32);
                    sy += __shfl_xor(sy, off, 32);
                }
                if (lrow == 0) {
                    int row = m * 16 + 8 * lhi + v;
                    atomicAdd(&pdsum[row * 2 + 0], sx);
                    atomicAdd(&pdsum[row * 2 + 1], sy);
                }
            }
        __syncthreads();   // pdsum complete

        // wave 0: clamp delta, integrate position, emit output
        if (wid == 0) {
            int row = lane;
            float dx = pdsum[row * 2 + 0] + bo0;
            float dy = pdsum[row * 2 + 1] + bo1;
            float nrm = sqrtf(dx * dx + dy * dy);
            float s = fminf(max_step / fmaxf(nrm, 1e-6f), 1.0f);
            dx *= s; dy *= s;
            posx += dx; posy += dy;
            pd[row * 2 + 0] = dx; pd[row * 2 + 1] = dy;
            pdsum[row * 2 + 0] = 0.0f; pdsum[row * 2 + 1] = 0.0f;
            float2 o; o.x = posx; o.y = posy;
            *(float2*)(out + ((size_t)(rowBase + row) * T + t) * 2) = o;
        }
        __syncthreads();
    }
}

extern "C" void kernel_launch(void* const* d_in, const int* in_sizes, int n_in,
                              void* d_out, int out_size, void* d_ws, size_t ws_size,
                              hipStream_t stream) {
    const float* init_h = (const float*)d_in[0];
    const float* ctx    = (const float*)d_in[1];
    const float* x0     = (const float*)d_in[2];
    const float* y0     = (const float*)d_in[3];
    const float* W_ih   = (const float*)d_in[4];
    const float* W_hh   = (const float*)d_in[5];
    const float* b_ih   = (const float*)d_in[6];
    const float* b_hh   = (const float*)d_in[7];
    const float* W_out  = (const float*)d_in[8];
    const float* b_out  = (const float*)d_in[9];
    const int*   Tp     = (const int*)d_in[10];
    float* out = (float*)d_out;
    unsigned short* whh_bf = (unsigned short*)d_ws;

    const int nW = 3 * HID * HID;  // 196608
    hipLaunchKernelGGL(whh_to_bf16, dim3((nW + 255) / 256), dim3(256), 0, stream,
                       W_hh, whh_bf, nW);

    const int N = in_sizes[2];     // 16384 rows
    hipLaunchKernelGGL(gru_rollout, dim3(N / 32), dim3(256), 0, stream,
                       init_h, ctx, x0, y0, W_ih, whh_bf, b_ih, b_hh,
                       W_out, b_out, Tp, out);
}
